// DAGN_HybridModel_39487929319657
// MI455X (gfx1250) — matrix operand order repaired and checked by reading, not compiled, not run on hardware
//
#include <hip/hip_runtime.h>
#include <hip/hip_bf16.h>

typedef __attribute__((ext_vector_type(16))) _Float16 v16h;
typedef __attribute__((ext_vector_type(8)))  float    v8f;

#define DEV __device__ __forceinline__

namespace {

constexpr int Bc = 32, Pc = 512, BSc = 64, Lc = 48, Hc = 128, NHc = 8, DHc = 16;
constexpr int NPc  = Bc * Pc;    // 16384 protein nodes
constexpr int NBSc = Bc * BSc;   // 2048 binding-site nodes
constexpr int NLGc = Bc * Lc;    // 1536 ligand nodes
constexpr int EPc  = NPc * 16;   // 262144 protein edges
constexpr int EBSc = NBSc * 12;  // 24576 bs edges
constexpr int ELc  = NLGc * 4;   // 6144 ligand edges
constexpr int ECc  = 16384;      // edge-chunk size (L2-resident pipeline)

DEV float fsilu(float v) { return v / (1.f + __expf(-v)); }
DEV float fsigm(float v) { return 1.f / (1.f + __expf(-v)); }

template<int ACT> DEV float actf(float v) {
  if (ACT == 1) return v > 0.f ? v : 0.f;   // relu
  if (ACT == 2) return fsilu(v);            // silu
  return v;
}

DEV float wsum(float v) {
#pragma unroll
  for (int o = 16; o; o >>= 1) v += __shfl_xor(v, o, 32);
  return v;
}
DEV float wmax(float v) {
#pragma unroll
  for (int o = 16; o; o >>= 1) v = fmaxf(v, __shfl_xor(v, o, 32));
  return v;
}

DEV void cvt8(v16h& a, float4 x, float4 y, int base) {
  a[base + 0] = (_Float16)x.x; a[base + 1] = (_Float16)x.y;
  a[base + 2] = (_Float16)x.z; a[base + 3] = (_Float16)x.w;
  a[base + 4] = (_Float16)y.x; a[base + 5] = (_Float16)y.y;
  a[base + 6] = (_Float16)y.z; a[base + 7] = (_Float16)y.w;
}

// ---------------------------------------------------------------------------
// Weight pre-pack: W[K,N] f32 (row-major) -> f16 fragments in this file's
// B-operand lane order:  Wp[((nt*Ks + ks)*32 + lane)*16 + e]
// where nt = 16-col tile, ks = 32-row K-step, and (contiguous k, same order as A) the lane
// needs element K = 32*ks + 16*(lane>>4) + e at column 16*nt + (lane&15).
// ---------------------------------------------------------------------------
__global__ void k_packw(const float* __restrict__ W, _Float16* __restrict__ Wp,
                        int N, int K) {
  int i = blockIdx.x * blockDim.x + threadIdx.x;   // K*N threads
  if (i >= K * N) return;
  int e    = i & 15;
  int lane = (i >> 4) & 31;
  int rest = i >> 9;
  int Ks   = K >> 5;
  int ks   = rest % Ks;
  int nt   = rest / Ks;
  int kk   = (ks << 5) + ((lane >> 4) << 4) + e;
  int nc   = (nt << 4) + (lane & 15);
  Wp[i] = (_Float16)W[(size_t)kk * N + nc];
}

// ---------------------------------------------------------------------------
// WMMA GEMM: C[M,N] = act(A[M,K] @ W[K,N] + bias [+ C]). A in f32 (converted on
// the fly), W pre-packed f16 fragments, f32 accumulate.
// One wave computes a 16x64 strip (4 WMMA tiles per K-step). All 12 loads of
// a K-step (4x b128 A + 4x 32B B fragments) are issued up front so the
// scheduler can stage partial loadcnt waits and overlap loads with WMMAs.
// REQUIRES K%32==0, N%64==0, M%16==0, 16B alignment.
// ---------------------------------------------------------------------------
template<int ACT, bool ACC, bool BIAS>
__global__ __launch_bounds__(256) void k_gemm(const float* __restrict__ A,
                                              const _Float16* __restrict__ Wp,
                                              const float* __restrict__ bias,
                                              float* __restrict__ C,
                                              int M, int N, int K) {
  const int wave  = blockIdx.x * (blockDim.x >> 5) + (threadIdx.x >> 5);
  const int Nt    = N >> 6;                    // 64-wide strips
  const int tiles = (M >> 4) * Nt;
  if (wave >= tiles) return;                   // wave-uniform: EXEC all-1s
  const int tm   = (wave / Nt) << 4;
  const int tnT  = (wave % Nt) << 2;           // first 16-col tile index
  const int lane = threadIdx.x & 31;
  const int half = lane >> 4;
  const int Ks   = K >> 5;

  v8f c[4] = {{}, {}, {}, {}};
  const float* arow = A + (size_t)(tm + (lane & 15)) * K + (half << 4);
  const _Float16* wb = Wp + (size_t)lane * 16;
  for (int ks = 0; ks < Ks; ++ks) {
    const int k0 = ks << 5;
    // issue ALL loads for this K-step first (A: 4x b128, B: 4x 2x b128)
    float4 a0 = *(const float4*)(arow + k0);
    float4 a1 = *(const float4*)(arow + k0 + 4);
    float4 a2 = *(const float4*)(arow + k0 + 8);
    float4 a3 = *(const float4*)(arow + k0 + 12);
    v16h bfrag[4];
#pragma unroll
    for (int nt = 0; nt < 4; ++nt)
      bfrag[nt] = *(const v16h*)(wb + ((size_t)(tnT + nt) * Ks + ks) * 512);
    // convert A while B fragments are in flight
    v16h a;
    cvt8(a, a0, a1, 0);
    cvt8(a, a2, a3, 8);
#pragma unroll
    for (int nt = 0; nt < 4; ++nt)
      c[nt] = __builtin_amdgcn_wmma_f32_16x16x32_f16(false, a, false, bfrag[nt],
                                                     (short)0, c[nt], false,
                                                     false);
  }
#pragma unroll
  for (int nt = 0; nt < 4; ++nt) {
#pragma unroll
    for (int r = 0; r < 8; ++r) {
      int row  = tm + r + (half << 3);         // C layout: M = r + 8*half
      int colv = ((tnT + nt) << 4) + (lane & 15);
      float v = c[nt][r];
      if (BIAS) v += bias[colv];
      size_t idx = (size_t)row * N + colv;
      if (ACC) v += C[idx];
      C[idx] = actf<ACT>(v);
    }
  }
}

template<int ACT, bool ACC, bool BIAS>
static void gemm(hipStream_t s, const float* A, const _Float16* Wp,
                 const float* b, float* C, int M, int N, int K) {
  int tiles = (M / 16) * (N / 64);
  k_gemm<ACT, ACC, BIAS><<<(tiles + 7) / 8, 256, 0, s>>>(A, Wp, b, C, M, N, K);
}

// ------------------------------- elementwise -------------------------------
__global__ void k_fill(float* p, size_t n, float v) {
  size_t i = (size_t)blockIdx.x * blockDim.x + threadIdx.x;
  if (i < n) p[i] = v;
}
__global__ void k_copy(float* dst, const float* src, size_t n) {
  size_t i = (size_t)blockIdx.x * blockDim.x + threadIdx.x;
  if (i < n) dst[i] = src[i];
}
__global__ void k_add(float* a, const float* b, size_t n) {
  size_t i = (size_t)blockIdx.x * blockDim.x + threadIdx.x;
  if (i < n) a[i] += b[i];
}
#define ELW(kern, n, ...) kern<<<(unsigned)(((n) + 255) / 256), 256, 0, stream>>>(__VA_ARGS__)

// -------------------------- feature construction ---------------------------
__global__ void k_embed(const float* ETab, const int* elem, float* ee,
                        float* featfull, int N) {
  int i = blockIdx.x * blockDim.x + threadIdx.x;   // N*16
  if (i >= N * 16) return;
  int n = i >> 4, f = i & 15;
  float v = ETab[elem[n] * 16 + f];
  ee[i] = v;
  featfull[(size_t)n * 64 + 48 + f] = v;           // concat tail
}
__global__ void k_ff48(const float* xff, float* featfull, int N) {
  int i = blockIdx.x * blockDim.x + threadIdx.x;   // N*48
  if (i >= N * 48) return;
  int n = i / 48, f = i % 48;
  featfull[(size_t)n * 64 + f] = xff[i];
}
// feat_bs padded to 64 cols: [xfc(32) | ee(16) | zeros(16)]
__global__ void k_featbs(const float* xfc, const float* ee, const int* bsidx,
                         float* featbs) {
  int i = blockIdx.x * blockDim.x + threadIdx.x;   // 2048*64
  if (i >= NBSc * 64) return;
  int n = i >> 6, f = i & 63;
  int g = bsidx[n];
  float v = 0.f;
  if (f < 32) v = xfc[(size_t)g * 32 + f];
  else if (f < 48) v = ee[(size_t)g * 16 + (f - 32)];
  featbs[i] = v;
}
// pad a [R0,Ncols] weight to [R1,Ncols] with zero rows
__global__ void k_padw(const float* w, float* wpad, int R0, int R1, int Ncols) {
  int i = blockIdx.x * blockDim.x + threadIdx.x;
  if (i >= R1 * Ncols) return;
  int r = i / Ncols;
  wpad[i] = (r < R0) ? w[i] : 0.f;
}
__global__ void k_gather3(const float* src, const int* idxv, float* dst, int n) {
  int i = blockIdx.x * blockDim.x + threadIdx.x;   // n*3
  if (i >= n * 3) return;
  dst[i] = src[(size_t)idxv[i / 3] * 3 + (i % 3)];
}
__global__ void k_role(const float* ca, const float* sc, const int* roles,
                       float* h, float* hinit) {
  int i = blockIdx.x * blockDim.x + threadIdx.x;   // NP*128
  if (i >= NPc * Hc) return;
  float v = (roles[i >> 7] == 0) ? ca[i] : sc[i];
  h[i] = v;
  hinit[i] = v;
}
__global__ void k_inject(float* h, const float* p_upd, const float* gate,
                         const int* bsidx) {
  int i = blockIdx.x * blockDim.x + threadIdx.x;   // 2048*128
  if (i >= NBSc * Hc) return;
  int e = i >> 7, f = i & 127;
  h[(size_t)bsidx[e] * Hc + f] += p_upd[i] * gate[e >> 6];
}

// ------------------------------- EGCL pieces -------------------------------
__global__ void k_edge_cd(const float* x, const int* row, const int* col,
                          int e0, int ec, float* cd, float* radial) {
  int e = blockIdx.x * blockDim.x + threadIdx.x;
  if (e >= ec) return;
  int r = row[e0 + e], c = col[e0 + e];
  float d0 = x[r * 3 + 0] - x[c * 3 + 0];
  float d1 = x[r * 3 + 1] - x[c * 3 + 1];
  float d2 = x[r * 3 + 2] - x[c * 3 + 2];
  cd[e * 3 + 0] = d0; cd[e * 3 + 1] = d1; cd[e * 3 + 2] = d2;
  radial[e] = d0 * d0 + d1 * d1 + d2 * d2;
}
__global__ void k_edge_gather(const float* h, const int* row, const int* col,
                              int e0, int ec, float* Hrow, float* Hcol) {
  int i = blockIdx.x * blockDim.x + threadIdx.x;   // ec*128
  if (i >= ec * Hc) return;
  int e = i >> 7, f = i & 127;
  Hrow[i] = h[(size_t)row[e0 + e] * Hc + f];
  Hcol[i] = h[(size_t)col[e0 + e] * Hc + f];
}
// m1 = silu(m1 + radial * ew1[row 256])
__global__ void k_edge_m1(float* m1, const float* radial, const float* w1c, int ec) {
  int i = blockIdx.x * blockDim.x + threadIdx.x;
  if (i >= ec * Hc) return;
  m1[i] = fsilu(m1[i] + radial[i >> 7] * w1c[i & 127]);
}
// m *= sigmoid(m . aw + ab)   (one wave per edge)
__global__ void k_edge_att(float* m, const float* aw, const float* ab, int ec) {
  int w = blockIdx.x * (blockDim.x >> 5) + (threadIdx.x >> 5);
  if (w >= ec) return;
  int lane = threadIdx.x & 31;
  float vals[4], s = 0.f;
#pragma unroll
  for (int i = 0; i < 4; ++i) {
    int f = lane + 32 * i;
    vals[i] = m[(size_t)w * Hc + f];
    s += vals[i] * aw[f];
  }
  s = wsum(s);
  float g = fsigm(s + ab[0]);
#pragma unroll
  for (int i = 0; i < 4; ++i) m[(size_t)w * Hc + (lane + 32 * i)] = vals[i] * g;
}
// t = tanh(tmp . cw2)   (no bias)
__global__ void k_edge_t(const float* tmp, const float* cw2, float* t, int ec) {
  int w = blockIdx.x * (blockDim.x >> 5) + (threadIdx.x >> 5);
  if (w >= ec) return;
  int lane = threadIdx.x & 31;
  float s = 0.f;
#pragma unroll
  for (int i = 0; i < 4; ++i) {
    int f = lane + 32 * i;
    s += tmp[(size_t)w * Hc + f] * cw2[f];
  }
  s = wsum(s);
  if (lane == 0) t[w] = tanhf(s);
}
__global__ void k_scatter_x(const float* cd, const float* t, const int* row,
                            int e0, int ec, float* xacc) {
  int e = blockIdx.x * blockDim.x + threadIdx.x;
  if (e >= ec) return;
  int r = row[e0 + e];
  float tv = t[e];
  atomicAdd(&xacc[r * 3 + 0], cd[e * 3 + 0] * tv);
  atomicAdd(&xacc[r * 3 + 1], cd[e * 3 + 1] * tv);
  atomicAdd(&xacc[r * 3 + 2], cd[e * 3 + 2] * tv);
}
__global__ void k_scatter_agg(const float* m, const int* row, int e0, int ec,
                              float* agg) {
  int i = blockIdx.x * blockDim.x + threadIdx.x;   // ec*128
  if (i >= ec * Hc) return;
  atomicAdd(&agg[(size_t)row[e0 + (i >> 7)] * Hc + (i & 127)], m[i]);
}
__global__ void k_count(const int* row, int E, float* cnt) {
  int e = blockIdx.x * blockDim.x + threadIdx.x;
  if (e >= E) return;
  atomicAdd(&cnt[row[e]], 1.f);
}
__global__ void k_coord_apply(float* x, const float* xacc, const float* cnt, int N) {
  int i = blockIdx.x * blockDim.x + threadIdx.x;   // N*3
  if (i >= N * 3) return;
  x[i] += xacc[i] / fmaxf(cnt[i / 3], 1.f);
}

// ----------------------------- attention / LN ------------------------------
// one wave per (batch, head, query); f32 VALU, shuffle reductions
__global__ void k_attn(const float* __restrict__ Q, const float* __restrict__ K,
                       const float* __restrict__ V, float* __restrict__ O,
                       int Bn, int Sq, int Sk) {
  int gw = blockIdx.x * (blockDim.x >> 5) + (threadIdx.x >> 5);
  if (gw >= Bn * NHc * Sq) return;
  int q = gw % Sq, t = gw / Sq, hh = t % NHc, b = t / NHc;
  int lane = threadIdx.x & 31;
  const float* qp = Q + ((size_t)(b * Sq + q) * Hc) + hh * DHc;
  float qr[DHc];
#pragma unroll
  for (int d = 0; d < DHc; ++d) qr[d] = qp[d];

  float sv[16];                 // Sk <= 512 -> <= 16 per lane
  int cnt = 0;
  float mx = -1e30f;
  for (int j = lane; j < Sk; j += 32) {
    const float* kp = K + ((size_t)(b * Sk + j) * Hc) + hh * DHc;
    float s = 0.f;
#pragma unroll
    for (int d = 0; d < DHc; ++d) s += qr[d] * kp[d];
    s *= 0.25f;                 // 1/sqrt(16)
    sv[cnt++] = s;
    mx = fmaxf(mx, s);
  }
  mx = wmax(mx);
  float sum = 0.f;
  for (int i = 0; i < cnt; ++i) { sv[i] = __expf(sv[i] - mx); sum += sv[i]; }
  sum = wsum(sum);
  float acc[DHc];
#pragma unroll
  for (int d = 0; d < DHc; ++d) acc[d] = 0.f;
  cnt = 0;
  for (int j = lane; j < Sk; j += 32) {
    const float* vp = V + ((size_t)(b * Sk + j) * Hc) + hh * DHc;
    float p = sv[cnt++];
#pragma unroll
    for (int d = 0; d < DHc; ++d) acc[d] += p * vp[d];
  }
#pragma unroll
  for (int d = 0; d < DHc; ++d) {
    float a = wsum(acc[d]);
    if (lane == 0) O[((size_t)(b * Sq + q) * Hc) + hh * DHc + d] = a / sum;
  }
}

__global__ void k_ln(float* X, const float* g, const float* bta, int Nrows) {
  int row = blockIdx.x * (blockDim.x >> 5) + (threadIdx.x >> 5);
  if (row >= Nrows) return;
  int lane = threadIdx.x & 31;
  float v[4], s = 0.f, s2 = 0.f;
#pragma unroll
  for (int i = 0; i < 4; ++i) {
    v[i] = X[(size_t)row * Hc + lane + 32 * i];
    s += v[i]; s2 += v[i] * v[i];
  }
  s = wsum(s); s2 = wsum(s2);
  float mean = s * (1.f / Hc);
  float var  = s2 * (1.f / Hc) - mean * mean;
  float rstd = rsqrtf(var + 1e-5f);
#pragma unroll
  for (int i = 0; i < 4; ++i) {
    int f = lane + 32 * i;
    X[(size_t)row * Hc + f] = (v[i] - mean) * rstd * g[f] + bta[f];
  }
}

__global__ void k_rowmean(const float* X, float* out, int Bn, int S) {
  int i = blockIdx.x * blockDim.x + threadIdx.x;   // Bn*128
  if (i >= Bn * Hc) return;
  int b = i >> 7, f = i & 127;
  float s = 0.f;
  for (int j = 0; j < S; ++j) s += X[((size_t)b * S + j) * Hc + f];
  out[i] = s / (float)S;
}

__global__ void k_head_out(const float* T, const float* W2, const float* b2,
                           float* out, float* gate, int Bn, int outd) {
  int i = blockIdx.x * blockDim.x + threadIdx.x;   // Bn*outd
  if (i >= Bn * outd) return;
  int b = i / outd, j = i % outd;
  float s = b2[j];
  for (int k = 0; k < Hc; ++k) s += T[(size_t)b * Hc + k] * W2[k * outd + j];
  out[i] = s;
  if (gate && j == 0) gate[b] = fsigm(s);
}

// ------------------------------ EGCL driver --------------------------------
struct Egcl {       // raw f32 params (biases / vectors used directly)
  const float *ew1, *eb1, *ew2, *eb2, *aw, *ab, *cw1, *cb1, *cw2,
              *nw1, *nb1, *nw2, *nb2;
};
struct EgclPk {     // packed f16 B-fragments for the GEMM-shaped weights
  const _Float16 *ew1a, *ew1b, *ew2, *cw1, *nw1a, *nw1b, *nw2;
};

static void run_egcl(hipStream_t stream, float* h, float* x, const int* rowp,
                     const int* colp, int E, int Nn, const Egcl& L,
                     const EgclPk& P, float* agg, float* xacc, const float* cnt,
                     float* eA, float* eB, float* eM, float* cd,
                     float* radial, float* tvals, float* tmpA) {
  ELW(k_fill, (size_t)Nn * Hc, agg, (size_t)Nn * Hc, 0.f);
  ELW(k_fill, (size_t)Nn * 3, xacc, (size_t)Nn * 3, 0.f);
  for (int e0 = 0; e0 < E; e0 += ECc) {
    int ec = (E - e0 < ECc) ? (E - e0) : ECc;
    ELW(k_edge_cd, ec, x, rowp, colp, e0, ec, cd, radial);
    ELW(k_edge_gather, ec * Hc, h, rowp, colp, e0, ec, eA, eB);
    // m1 = silu(Hrow@W1a + Hcol@W1b + radial*w1c + eb1)
    gemm<0, false, true >(stream, eA, P.ew1a, L.eb1,   eM, ec, Hc, Hc);
    gemm<0, true,  false>(stream, eB, P.ew1b, nullptr, eM, ec, Hc, Hc);
    ELW(k_edge_m1, ec * Hc, eM, radial, L.ew1 + 256 * Hc, ec);
    // m = silu(m1@ew2 + eb2) * sigmoid(m@aw + ab)
    gemm<2, false, true>(stream, eM, P.ew2, L.eb2, eA, ec, Hc, Hc);
    k_edge_att<<<(ec + 3) / 4, 128, 0, stream>>>(eA, L.aw, L.ab, ec);
    // t = tanh(silu(m@cw1 + cb1) @ cw2)
    gemm<2, false, true>(stream, eA, P.cw1, L.cb1, eB, ec, Hc, Hc);
    k_edge_t<<<(ec + 3) / 4, 128, 0, stream>>>(eB, L.cw2, tvals, ec);
    ELW(k_scatter_x, ec, cd, tvals, rowp, e0, ec, xacc);
    ELW(k_scatter_agg, ec * Hc, eA, rowp, e0, ec, agg);
  }
  ELW(k_coord_apply, Nn * 3, x, xacc, cnt, Nn);
  // h += silu([h,agg]@nw1 + nb1) @ nw2 + nb2
  gemm<0, false, true >(stream, h,    P.nw1a, L.nb1,   tmpA, Nn, Hc, Hc);
  gemm<2, true,  false>(stream, agg,  P.nw1b, nullptr, tmpA, Nn, Hc, Hc);
  gemm<0, true,  true >(stream, tmpA, P.nw2,  L.nb2,   h,    Nn, Hc, Hc);
}

} // namespace

// ===========================================================================
extern "C" void kernel_launch(void* const* d_in, const int* in_sizes, int n_in,
                              void* d_out, int out_size, void* d_ws, size_t ws_size,
                              hipStream_t stream) {
  (void)in_sizes; (void)n_in; (void)out_size; (void)ws_size;

  // ---- parameter walk: setup_inputs() nested-dict insertion order ----
  int ip = 0;
  auto F = [&]() { return (const float*)d_in[ip++]; };
  auto I = [&]() { return (const int*)d_in[ip++]; };
  auto egcl_in = [&](Egcl& L) {
    L.ew1 = F(); L.eb1 = F(); L.ew2 = F(); L.eb2 = F(); L.aw = F(); L.ab = F();
    L.cw1 = F(); L.cb1 = F(); L.cw2 = F(); L.nw1 = F(); L.nb1 = F();
    L.nw2 = F(); L.nb2 = F();
  };

  const float* elem_tab = F();                               // params.elem_emb
  const float *bs_win = F(), *bs_bin = F();                  // bs_enc
  Egcl bsL[4]; for (int l = 0; l < 4; ++l) egcl_in(bsL[l]);
  const float *bs_wout = F(), *bs_bout = F();
  const float *lg_win = F(), *lg_bin = F();                  // lig_enc
  Egcl lgL[4]; for (int l = 0; l < 4; ++l) egcl_in(lgL[l]);
  const float *lg_wout = F(), *lg_bout = F();
  const float* p2l[8]; for (int j = 0; j < 8; ++j) p2l[j] = F(); // wq,bq,wk,bk,wv,bv,wo,bo
  const float* l2p[8]; for (int j = 0; j < 8; ++j) l2p[j] = F();
  const float* bindp[4]; for (int j = 0; j < 4; ++j) bindp[j] = F(); // w1,b1,w2,b2
  const float* actp[4];  for (int j = 0; j < 4; ++j) actp[j]  = F();
  const float *wca = F(), *bca = F(), *wsc = F(), *bsc = F();
  Egcl prL[4]; for (int l = 0; l < 4; ++l) egcl_in(prL[l]);  // prop_enc.layers
  const float *pr_wout = F(), *pr_bout = F();
  const float *fng = F(), *fnb = F();
  const float* tx[2][16];                                    // attn(8)+ln1g,ln1b,w1,b1,w2,b2,ln2g,ln2b
  for (int t = 0; t < 2; ++t) for (int j = 0; j < 16; ++j) tx[t][j] = F();

  const float* xfc    = F();   // x_float_clean [16384,32]
  const float* xff    = F();   // x_float_full  [16384,48]
  const float* ppos   = F();   // p_pos [16384,3]
  const float* ligx   = F();   // lig_x [1536,32]
  const float* ligpos = F();   // lig_pos [1536,3]
  const int* xelem  = I();
  const int* roles  = I();
  const int* pedge  = I();     // [2, 262144]
  const int* bsedge = I();     // [2, 24576]
  const int* ledge  = I();     // [2, 6144]
  const int* bsidx  = I();     // [2048]
  const int* pbatch = I(); (void)pbatch;
  const int* lbatch = I(); (void)lbatch;

  // ---- workspace arena (floats; every block a multiple of 16 floats) ----
  float* base = (float*)d_ws;
  size_t off = 0;
  auto A_ = [&](size_t n) { float* p = base + off; off += n; return p; };
  float* ee       = A_((size_t)NPc * 16);
  float* featfull = A_((size_t)NPc * 64);
  float* featbs   = A_((size_t)NBSc * 64);   // padded to 64 cols (K%32==0)
  float* win_pad  = A_(64 * Hc);             // bs_enc.win zero-padded 48->64 rows
  float* h_bs     = A_((size_t)NBSc * Hc);
  float* x_bs     = A_((size_t)NBSc * 3);
  float* hbs_out  = A_((size_t)NBSc * Hc);
  float* h_l      = A_((size_t)NLGc * Hc);
  float* x_l      = A_((size_t)NLGc * 3);
  float* hl_out   = A_((size_t)NLGc * Hc);
  float* qb       = A_((size_t)NBSc * Hc);
  float* kb       = A_((size_t)NBSc * Hc);
  float* vb       = A_((size_t)NBSc * Hc);
  float* ob       = A_((size_t)NBSc * Hc);
  float* p_upd    = A_((size_t)NBSc * Hc);
  float* l_upd    = A_((size_t)NLGc * Hc);
  float* p_vec    = A_(Bc * Hc);
  float* l_vec    = A_(Bc * Hc);
  float* headtmp  = A_(Bc * Hc);
  float* gate     = A_(Bc);
  float* h        = A_((size_t)NPc * Hc);
  float* h_init   = A_((size_t)NPc * Hc);
  float* xtx      = A_((size_t)NPc * Hc);
  float* coords   = A_((size_t)NPc * 3);
  float* xacc     = A_((size_t)NPc * 3);
  float* cnt_p    = A_(NPc);
  float* cnt_bs   = A_(NBSc);
  float* cnt_l    = A_(NLGc);
  float* agg      = A_((size_t)NPc * Hc);
  float* tmpA     = A_((size_t)NPc * Hc);   // |
  float* tmpB     = A_((size_t)NPc * Hc);   // | contiguous: FFN scratch
  float* eA       = A_((size_t)ECc * Hc);   // | overlays tmpA..eB
  float* eB       = A_((size_t)ECc * Hc);   // | (4*2,097,152 = 16384*512)
  float* eM       = A_((size_t)ECc * Hc);
  float* cd       = A_((size_t)ECc * 3);
  float* radial   = A_(ECc);
  float* tvals    = A_(ECc);
  float* ffn      = tmpA;                   // 16384x512, used when edge bufs dead

  // packed-weight sub-arena (f16), filled once per launch
  auto packw = [&](const float* W, int K, int N) -> const _Float16* {
    _Float16* p = (_Float16*)A_((size_t)K * N / 2);
    int total = K * N;
    k_packw<<<(total + 255) / 256, 256, 0, stream>>>(W, p, N, K);
    return p;
  };

  float* outp = (float*)d_out;              // [0,32): bind, [32,96): act, [96,..): coords

  // ---- 0. pad + pack all GEMM weights to f16 WMMA fragments ----
  ELW(k_padw, 64 * Hc, bs_win, win_pad, 48, 64, Hc);
  const _Float16* P_win = packw(win_pad, 64, Hc);
  EgclPk bsP[4], lgP[4], prP[4];
  auto pack_egcl = [&](const Egcl& L, EgclPk& P) {
    P.ew1a = packw(L.ew1, Hc, Hc);
    P.ew1b = packw(L.ew1 + 128 * Hc, Hc, Hc);
    P.ew2  = packw(L.ew2, Hc, Hc);
    P.cw1  = packw(L.cw1, Hc, Hc);
    P.nw1a = packw(L.nw1, Hc, Hc);
    P.nw1b = packw(L.nw1 + 128 * Hc, Hc, Hc);
    P.nw2  = packw(L.nw2, Hc, Hc);
  };
  for (int l = 0; l < 4; ++l) pack_egcl(bsL[l], bsP[l]);
  for (int l = 0; l < 4; ++l) pack_egcl(lgL[l], lgP[l]);
  for (int l = 0; l < 4; ++l) pack_egcl(prL[l], prP[l]);
  const _Float16* P_bswout = packw(bs_wout, Hc, Hc);
  const _Float16* P_lgwin  = packw(lg_win, 32, Hc);
  const _Float16* P_lgwout = packw(lg_wout, Hc, Hc);
  const _Float16* P_p2l[4] = {packw(p2l[0], Hc, Hc), packw(p2l[2], Hc, Hc),
                              packw(p2l[4], Hc, Hc), packw(p2l[6], Hc, Hc)};
  const _Float16* P_l2p[4] = {packw(l2p[0], Hc, Hc), packw(l2p[2], Hc, Hc),
                              packw(l2p[4], Hc, Hc), packw(l2p[6], Hc, Hc)};
  const _Float16* P_bind1a = packw(bindp[0], Hc, Hc);
  const _Float16* P_bind1b = packw(bindp[0] + 128 * Hc, Hc, Hc);
  const _Float16* P_act1a  = packw(actp[0], Hc, Hc);
  const _Float16* P_act1b  = packw(actp[0] + 128 * Hc, Hc, Hc);
  const _Float16* P_wca    = packw(wca, 64, Hc);
  const _Float16* P_wsc    = packw(wsc, 64, Hc);
  const _Float16* P_prwout = packw(pr_wout, Hc, Hc);
  const _Float16* P_tx[2][6];
  for (int t = 0; t < 2; ++t) {
    P_tx[t][0] = packw(tx[t][0], Hc, Hc);    // wq
    P_tx[t][1] = packw(tx[t][2], Hc, Hc);    // wk
    P_tx[t][2] = packw(tx[t][4], Hc, Hc);    // wv
    P_tx[t][3] = packw(tx[t][6], Hc, Hc);    // wo
    P_tx[t][4] = packw(tx[t][10], Hc, 512);  // w1 [128,512]
    P_tx[t][5] = packw(tx[t][12], 512, Hc);  // w2 [512,128]
  }

  // ---- 1. embeddings & feature concat ----
  ELW(k_embed, NPc * 16, elem_tab, xelem, ee, featfull, NPc);
  ELW(k_ff48, NPc * 48, xff, featfull, NPc);
  ELW(k_featbs, NBSc * 64, xfc, ee, bsidx, featbs);
  ELW(k_gather3, NBSc * 3, ppos, bsidx, x_bs, NBSc);
  ELW(k_copy, (size_t)NLGc * 3, x_l, ligpos, (size_t)NLGc * 3);

  // ---- 2. binding-site EGNN ----
  ELW(k_fill, (size_t)NBSc, cnt_bs, (size_t)NBSc, 0.f);
  ELW(k_count, EBSc, bsedge, EBSc, cnt_bs);
  gemm<0, false, true>(stream, featbs, P_win, bs_bin, h_bs, NBSc, Hc, 64);
  for (int l = 0; l < 4; ++l)
    run_egcl(stream, h_bs, x_bs, bsedge, bsedge + EBSc, EBSc, NBSc, bsL[l],
             bsP[l], agg, xacc, cnt_bs, eA, eB, eM, cd, radial, tvals, tmpA);
  gemm<0, false, true>(stream, h_bs, P_bswout, bs_bout, hbs_out, NBSc, Hc, Hc);

  // ---- 3. ligand EGNN ----
  ELW(k_fill, (size_t)NLGc, cnt_l, (size_t)NLGc, 0.f);
  ELW(k_count, ELc, ledge, ELc, cnt_l);
  gemm<0, false, true>(stream, ligx, P_lgwin, lg_bin, h_l, NLGc, Hc, 32);
  for (int l = 0; l < 4; ++l)
    run_egcl(stream, h_l, x_l, ledge, ledge + ELc, ELc, NLGc, lgL[l],
             lgP[l], agg, xacc, cnt_l, eA, eB, eM, cd, radial, tvals, tmpA);
  gemm<0, false, true>(stream, h_l, P_lgwout, lg_bout, hl_out, NLGc, Hc, Hc);

  // ---- 4. cross attention (masks all-true) ----
  // p2l: Q from bs (Sq=64), K/V from lig (Sk=48)
  gemm<0, false, true>(stream, hbs_out, P_p2l[0], p2l[1], qb, NBSc, Hc, Hc);
  gemm<0, false, true>(stream, hl_out,  P_p2l[1], p2l[3], kb, NLGc, Hc, Hc);
  gemm<0, false, true>(stream, hl_out,  P_p2l[2], p2l[5], vb, NLGc, Hc, Hc);
  k_attn<<<(Bc * NHc * BSc + 3) / 4, 128, 0, stream>>>(qb, kb, vb, ob, Bc, BSc, Lc);
  gemm<0, false, true>(stream, ob, P_p2l[3], p2l[7], p_upd, NBSc, Hc, Hc);
  // l2p: Q from lig (Sq=48), K/V from bs (Sk=64)
  gemm<0, false, true>(stream, hl_out,  P_l2p[0], l2p[1], qb, NLGc, Hc, Hc);
  gemm<0, false, true>(stream, hbs_out, P_l2p[1], l2p[3], kb, NBSc, Hc, Hc);
  gemm<0, false, true>(stream, hbs_out, P_l2p[2], l2p[5], vb, NBSc, Hc, Hc);
  k_attn<<<(Bc * NHc * Lc + 3) / 4, 128, 0, stream>>>(qb, kb, vb, ob, Bc, Lc, BSc);
  gemm<0, false, true>(stream, ob, P_l2p[3], l2p[7], l_upd, NLGc, Hc, Hc);

  ELW(k_rowmean, Bc * Hc, p_upd, p_vec, Bc, BSc);
  ELW(k_rowmean, Bc * Hc, l_upd, l_vec, Bc, Lc);

  // ---- 5. binding head + gate ----
  gemm<0, false, true >(stream, p_vec, P_bind1a, bindp[1], headtmp, Bc, Hc, Hc);
  gemm<1, true,  false>(stream, l_vec, P_bind1b, nullptr,  headtmp, Bc, Hc, Hc);
  k_head_out<<<1, 64, 0, stream>>>(headtmp, bindp[2], bindp[3], outp, gate, Bc, 1);

  // ---- 6. full-protein node init ----
  gemm<0, false, true>(stream, featfull, P_wca, bca, tmpA, NPc, Hc, 64);
  gemm<0, false, true>(stream, featfull, P_wsc, bsc, tmpB, NPc, Hc, 64);
  ELW(k_role, NPc * Hc, tmpA, tmpB, roles, h, h_init);
  ELW(k_copy, (size_t)NPc * 3, coords, ppos, (size_t)NPc * 3);

  // ---- 7. propagation EGNN (gated injection each layer) ----
  ELW(k_fill, (size_t)NPc, cnt_p, (size_t)NPc, 0.f);
  ELW(k_count, EPc, pedge, EPc, cnt_p);
  for (int l = 0; l < 4; ++l) {
    ELW(k_inject, NBSc * Hc, h, p_upd, gate, bsidx);
    run_egcl(stream, h, coords, pedge, pedge + EPc, EPc, NPc, prL[l],
             prP[l], agg, xacc, cnt_p, eA, eB, eM, cd, radial, tvals, tmpA);
  }
  ELW(k_add, (size_t)NPc * Hc, h, h_init, (size_t)NPc * Hc);
  gemm<0, false, true>(stream, h, P_prwout, pr_bout, xtx, NPc, Hc, Hc);
  k_ln<<<(NPc + 3) / 4, 128, 0, stream>>>(xtx, fng, fnb, NPc);

  // ---- 8. transformer (post-norm, relu FFN) ----
  for (int t = 0; t < 2; ++t) {
    const float* const* w = tx[t];
    float *tq = agg, *tk = eA, *tv = eB, *ao = eM;   // reuse dead buffers
    gemm<0, false, true>(stream, xtx, P_tx[t][0], w[1], tq, NPc, Hc, Hc);
    gemm<0, false, true>(stream, xtx, P_tx[t][1], w[3], tk, NPc, Hc, Hc);
    gemm<0, false, true>(stream, xtx, P_tx[t][2], w[5], tv, NPc, Hc, Hc);
    k_attn<<<(Bc * NHc * Pc + 3) / 4, 128, 0, stream>>>(tq, tk, tv, ao, Bc, Pc, Pc);
    gemm<0, true, true>(stream, ao, P_tx[t][3], w[7], xtx, NPc, Hc, Hc); // x += attn@wo+bo
    k_ln<<<(NPc + 3) / 4, 128, 0, stream>>>(xtx, w[8], w[9], NPc);       // ln1
    gemm<1, false, true>(stream, xtx, P_tx[t][4], w[11], ffn, NPc, 512, Hc);
    gemm<0, true,  true>(stream, ffn, P_tx[t][5], w[13], xtx, NPc, Hc, 512); // x += ffn
    k_ln<<<(NPc + 3) / 4, 128, 0, stream>>>(xtx, w[14], w[15], NPc);     // ln2
  }

  // ---- 9. pooled activity head + outputs ----
  ELW(k_rowmean, Bc * Hc, xtx, p_vec, Bc, Pc);   // pooled (reuse p_vec)
  gemm<0, false, true >(stream, p_vec, P_act1a, actp[1], headtmp, Bc, Hc, Hc);
  gemm<1, true,  false>(stream, l_vec, P_act1b, nullptr, headtmp, Bc, Hc, Hc);
  k_head_out<<<1, 64, 0, stream>>>(headtmp, actp[2], actp[3], outp + 32, nullptr, Bc, 2);
  ELW(k_copy, (size_t)NPc * 3, outp + 96, coords, (size_t)NPc * 3);
}
